// MultiHeadAttention_9818295239193
// MI455X (gfx1250) — matrix operand order repaired and checked by reading, not compiled, not run on hardware
//
#include <hip/hip_runtime.h>

#define BB 4
#define SS 2048
#define DD 1024
#define HH 16
#define DKK 64

typedef __attribute__((ext_vector_type(16))) __bf16 v16bf;
typedef __attribute__((ext_vector_type(8)))  __bf16 v8bf;
typedef __attribute__((ext_vector_type(8)))  float  v8f;

// ---------------- helpers ----------------

__device__ __forceinline__ __bf16 cvtbf(float f) { return (__bf16)f; }

__device__ __forceinline__ v8f vzero8() {
  v8f z;
#pragma unroll
  for (int i = 0; i < 8; ++i) z[i] = 0.0f;
  return z;
}

__device__ __forceinline__ v8f wmma_bf16(v16bf a, v16bf b, v8f c) {
  // D = A(16x32 bf16) x B(32x16 bf16) + C(16x16 f32)
  return __builtin_amdgcn_wmma_f32_16x16x32_bf16(
      false, a, false, b, (short)0, c, false, false);
}

// A fragment (16x32, M = lane&15) from a bf16 row. Contiguous K layout:
// lanes 0-15: K {k0+0..15}; lanes 16-31: K {k0+16..31}
__device__ __forceinline__ v16bf a_frag_bf16(const __bf16* __restrict__ row,
                                             int k0, int lane) {
  const int b0 = k0 + ((lane & 16) ? 16 : 0);
  v16bf a;
  ((v8bf*)&a)[0] = *(const v8bf*)(row + b0);
  ((v8bf*)&a)[1] = *(const v8bf*)(row + b0 + 8);
  return a;
}

// Async 16B copy global -> LDS (CDNA5 ASYNCcnt path). LDS aperture flat
// addresses carry the LDS byte offset in addr[31:0] (ISA 10.2).
__device__ __forceinline__ void async_cp16(const void* gptr, void* lptr) {
  const unsigned lds = (unsigned)(unsigned long long)lptr;
  const unsigned long long ga = (unsigned long long)gptr;
  asm volatile("global_load_async_to_lds_b128 %0, %1, off"
               :: "v"(lds), "v"(ga) : "memory");
}

// ---------------- kernel 0: bulk fp32 -> bf16 conversion ----------------
__global__ __launch_bounds__(256) void cvt_f32_bf16_kernel(
    const float* __restrict__ in, __bf16* __restrict__ out, int n8) {
  const int i = blockIdx.x * 256 + threadIdx.x;
  if (i >= n8) return;
  const float4* p = (const float4*)in + 2 * (size_t)i;
  const float4 x0 = p[0];
  const float4 x1 = p[1];
  v8bf o;
  o[0] = cvtbf(x0.x); o[1] = cvtbf(x0.y); o[2] = cvtbf(x0.z); o[3] = cvtbf(x0.w);
  o[4] = cvtbf(x1.x); o[5] = cvtbf(x1.y); o[6] = cvtbf(x1.z); o[7] = cvtbf(x1.w);
  *((v8bf*)out + i) = o;
}

// ---------------- kernel 1: QKV projection (bf16 GEMM, bf16 out) ----------------
// Y = X @ W^T + bias ; per wave: 32(M) x 64(N) tile, B fragments reused 2x.
// mode 0: out [B,H,S,DK] (Q,K) ; mode 1: out [B,H,DK,S] (V^T)
__global__ __launch_bounds__(128) void gemm_qkv_kernel(
    const __bf16* __restrict__ X, const __bf16* __restrict__ W,
    const float* __restrict__ bias, __bf16* __restrict__ out, int mode) {
  const int lane  = threadIdx.x & 31;
  const int wv    = threadIdx.x >> 5;
  const int lm    = lane & 15;
  const int rbase = (lane & 16) ? 8 : 0;
  const int koff  = (lane & 16) ? 16 : 0;
  const int n0    = blockIdx.x * 64;             // output-column tile
  const int m0    = (blockIdx.y * 4 + wv) * 32;  // output-row tile (per wave)

  const __bf16* xrow0 = X + (size_t)(m0 + lm) * DD;
  const __bf16* xrow1 = xrow0 + (size_t)16 * DD;

  v8f acc[2][4];
#pragma unroll
  for (int u = 0; u < 2; ++u)
#pragma unroll
    for (int t = 0; t < 4; ++t) acc[u][t] = vzero8();

  for (int k0 = 0; k0 < DD; k0 += 32) {
    const v16bf a0 = a_frag_bf16(xrow0, k0, lane);
    const v16bf a1 = a_frag_bf16(xrow1, k0, lane);
#pragma unroll
    for (int t = 0; t < 4; ++t) {
      const __bf16* wp = W + (size_t)(n0 + t * 16 + lm) * DD + k0 + koff;
      const v16bf b = *(const v16bf*)wp;
      acc[0][t] = wmma_bf16(a0, b, acc[0][t]);
      acc[1][t] = wmma_bf16(a1, b, acc[1][t]);
    }
  }

  const int bidx = m0 / SS;
  const int h    = n0 / DKK;  // n0 multiple of 64 -> one head per wave

#pragma unroll
  for (int u = 0; u < 2; ++u) {
    const int s0 = (m0 % SS) + u * 16;
#pragma unroll
    for (int t = 0; t < 4; ++t) {
      const float bvl = bias[n0 + t * 16 + lm];
      const int   dk  = t * 16 + lm;
      if (mode == 0) {
        __bf16* op =
            out + (((size_t)bidx * HH + h) * SS + s0 + rbase) * DKK + dk;
#pragma unroll
        for (int r = 0; r < 8; ++r)
          op[(size_t)r * DKK] = cvtbf(acc[u][t][r] + bvl);
      } else {
        v8bf pk;
#pragma unroll
        for (int r = 0; r < 8; ++r) pk[r] = cvtbf(acc[u][t][r] + bvl);
        *(v8bf*)(out + (((size_t)bidx * HH + h) * DKK + dk) * SS + s0 + rbase) =
            pk;
      }
    }
  }
}

// ---------------- kernel 2: flash attention ----------------
// Q,K: [B*H, S, DK] bf16 ; Vt: [B*H, DK, S] bf16 ; Xatt: [B*S, D] bf16
// Per wave: 32 query rows (2 subtiles). K/V tiles are staged to LDS once per
// workgroup with global_load_async_to_lds_b128 (double-buffered, ASYNCcnt).
__global__ __launch_bounds__(128) void flash_attn_kernel(
    const __bf16* __restrict__ Q, const __bf16* __restrict__ K,
    const __bf16* __restrict__ Vt, __bf16* __restrict__ Xatt) {
  __shared__ __bf16 pbuf[4][2][16][40];  // per-wave 2x (16x32) P tiles, padded
  __shared__ __bf16 kbuf[2][32][64];     // [buf][key row][dk]     (4 KB each)
  __shared__ __bf16 vbuf[2][64][32];     // [buf][dk][key in tile] (4 KB each)

  const int tid   = threadIdx.x;
  const int lane  = tid & 31;
  const int wv    = tid >> 5;
  const int lm    = lane & 15;
  const int rbase = (lane & 16) ? 8 : 0;
  const int koff  = (lane & 16) ? 16 : 0;
  const int bh    = blockIdx.y;
  const int q0    = (blockIdx.x * 4 + wv) * 32;
  // softmax in log2 domain: fold 1/sqrt(64) * log2(e) into one multiplier
  const float sl2e = 0.125f * 1.44269504088896340736f;

  const __bf16* Kbase  = K + (size_t)bh * SS * DKK;   // rows of 64 bf16
  const __bf16* Vtbase = Vt + (size_t)bh * DKK * SS;  // rows of SS bf16

  v16bf qa[2][2];
#pragma unroll
  for (int u = 0; u < 2; ++u) {
    const __bf16* qrow = Q + ((size_t)bh * SS + q0 + u * 16 + lm) * DKK;
    qa[u][0] = a_frag_bf16(qrow, 0, lane);
    qa[u][1] = a_frag_bf16(qrow, 32, lane);
  }

  float mrow[2][8], lrow[2][8];
  v8f o[2][4];
#pragma unroll
  for (int u = 0; u < 2; ++u) {
#pragma unroll
    for (int r = 0; r < 8; ++r) { mrow[u][r] = -3.0e38f; lrow[u][r] = 0.0f; }
#pragma unroll
    for (int t = 0; t < 4; ++t) o[u][t] = vzero8();
  }

  // stage key-block kb (32 keys) into LDS buffer `buf` (128 threads, 16B each)
  auto stage = [&](int kstart, int buf) {
#pragma unroll
    for (int i = 0; i < 2; ++i) {
      const int c  = tid + i * 128;          // chunk id 0..255 (16B chunks)
      const int kr = c >> 3, kc = (c & 7) * 8;   // K: 32 rows x 128B
      async_cp16(Kbase + (size_t)(kstart + kr) * DKK + kc, &kbuf[buf][kr][kc]);
      const int vr = c >> 2, vc = (c & 3) * 8;   // Vt: 64 rows x 64B
      async_cp16(Vtbase + (size_t)vr * SS + kstart + vc, &vbuf[buf][vr][vc]);
    }
  };

  stage(0, 0);  // prologue: first tile in flight

  for (int it = 0; it < SS / 32; ++it) {
    asm volatile("s_wait_asynccnt 0x0" ::: "memory");  // own copies of tile it
    __syncthreads();  // tile `it` visible; buffer (it+1)&1 free for reuse
    if (it + 1 < SS / 32) stage((it + 1) * 32, (it + 1) & 1);

    const __bf16(*kb)[64] = kbuf[it & 1];
    const __bf16(*vb)[32] = vbuf[it & 1];

    // ---- scores S = Q * K^T for 32 keys, K fragments from LDS ----
    const v16bf b00 = *(const v16bf*)(&kb[lm][koff]);
    const v16bf b01 = *(const v16bf*)(&kb[lm][32 + koff]);
    const v16bf b10 = *(const v16bf*)(&kb[16 + lm][koff]);
    const v16bf b11 = *(const v16bf*)(&kb[16 + lm][32 + koff]);

    v8f sc[2][2];
#pragma unroll
    for (int u = 0; u < 2; ++u) {
      sc[u][0] = vzero8();
      sc[u][1] = vzero8();
      sc[u][0] = wmma_bf16(qa[u][0], b00, sc[u][0]);
      sc[u][0] = wmma_bf16(qa[u][1], b01, sc[u][0]);
      sc[u][1] = wmma_bf16(qa[u][0], b10, sc[u][1]);
      sc[u][1] = wmma_bf16(qa[u][1], b11, sc[u][1]);
    }

    // ---- online softmax (rows live in one half-wave; xor<16 stays in half) ----
#pragma unroll
    for (int u = 0; u < 2; ++u) {
      float corr[8];
#pragma unroll
      for (int r = 0; r < 8; ++r) {
        const float a0 = sc[u][0][r] * sl2e;
        const float a1 = sc[u][1][r] * sl2e;
        float mx = fmaxf(a0, a1);
#pragma unroll
        for (int msk = 1; msk < 16; msk <<= 1)
          mx = fmaxf(mx, __shfl_xor(mx, msk, 32));
        const float mn = fmaxf(mrow[u][r], mx);
        corr[r] = exp2f(mrow[u][r] - mn);
        mrow[u][r] = mn;
        const float p0 = exp2f(a0 - mn);
        const float p1 = exp2f(a1 - mn);
        sc[u][0][r] = p0; sc[u][1][r] = p1;
        float rs = p0 + p1;
#pragma unroll
        for (int msk = 1; msk < 16; msk <<= 1) rs += __shfl_xor(rs, msk, 32);
        lrow[u][r] = lrow[u][r] * corr[r] + rs;
      }
#pragma unroll
      for (int t = 0; t < 4; ++t)
#pragma unroll
        for (int r = 0; r < 8; ++r) o[u][t][r] *= corr[r];

      // ---- transpose P (C layout -> A layout) through wave-private LDS ----
#pragma unroll
      for (int r = 0; r < 8; ++r) {
        pbuf[wv][u][rbase + r][lm]      = cvtbf(sc[u][0][r]);
        pbuf[wv][u][rbase + r][lm + 16] = cvtbf(sc[u][1][r]);
      }
    }
    asm volatile("s_wait_dscnt 0x0" ::: "memory");  // in-wave DS RAW

    v16bf pa[2];
#pragma unroll
    for (int u = 0; u < 2; ++u) {
      const __bf16* prow = &pbuf[wv][u][lm][0];
      const int b0 = (lane & 16) ? 16 : 0;
      ((v8bf*)&pa[u])[0] = *(const v8bf*)(prow + b0);
      ((v8bf*)&pa[u])[1] = *(const v8bf*)(prow + b0 + 8);
    }

    // ---- O += P * V  (V fragments from LDS, shared across subtiles) ----
#pragma unroll
    for (int t = 0; t < 4; ++t) {
      const v16bf vbf = *(const v16bf*)(&vb[t * 16 + lm][koff]);
      o[0][t] = wmma_bf16(pa[0], vbf, o[0][t]);
      o[1][t] = wmma_bf16(pa[1], vbf, o[1][t]);
    }
  }

  // ---- epilogue: normalize, write context to [B*S, D] bf16 ----
  const int b = bh / HH, h = bh % HH;
#pragma unroll
  for (int u = 0; u < 2; ++u) {
#pragma unroll
    for (int t = 0; t < 4; ++t) {
      __bf16* op = Xatt + ((size_t)b * SS + q0 + u * 16 + rbase) * DD +
                   h * DKK + t * 16 + lm;
#pragma unroll
      for (int r = 0; r < 8; ++r)
        op[(size_t)r * DD] = cvtbf(o[u][t][r] / lrow[u][r]);
    }
  }
}

// ---------------- kernel 3: output projection (bf16 GEMM, fp32 out) ----------------
__global__ __launch_bounds__(128) void gemm_out_kernel(
    const __bf16* __restrict__ Xatt, const __bf16* __restrict__ W,
    const float* __restrict__ bias, float* __restrict__ out) {
  const int lane  = threadIdx.x & 31;
  const int wv    = threadIdx.x >> 5;
  const int lm    = lane & 15;
  const int rbase = (lane & 16) ? 8 : 0;
  const int koff  = (lane & 16) ? 16 : 0;
  const int n0    = blockIdx.x * 64;
  const int m0    = (blockIdx.y * 4 + wv) * 32;

  const __bf16* xrow0 = Xatt + (size_t)(m0 + lm) * DD;
  const __bf16* xrow1 = xrow0 + (size_t)16 * DD;

  v8f acc[2][4];
#pragma unroll
  for (int u = 0; u < 2; ++u)
#pragma unroll
    for (int t = 0; t < 4; ++t) acc[u][t] = vzero8();

  for (int k0 = 0; k0 < DD; k0 += 32) {
    const v16bf a0 = a_frag_bf16(xrow0, k0, lane);
    const v16bf a1 = a_frag_bf16(xrow1, k0, lane);
#pragma unroll
    for (int t = 0; t < 4; ++t) {
      const __bf16* wp = W + (size_t)(n0 + t * 16 + lm) * DD + k0 + koff;
      const v16bf b = *(const v16bf*)wp;
      acc[0][t] = wmma_bf16(a0, b, acc[0][t]);
      acc[1][t] = wmma_bf16(a1, b, acc[1][t]);
    }
  }

#pragma unroll
  for (int u = 0; u < 2; ++u) {
#pragma unroll
    for (int t = 0; t < 4; ++t) {
      const float bvl = bias[n0 + t * 16 + lm];
      float* op = out + (size_t)(m0 + u * 16 + rbase) * DD + n0 + t * 16 + lm;
#pragma unroll
      for (int r = 0; r < 8; ++r) op[(size_t)r * DD] = acc[u][t][r] + bvl;
    }
  }
}

// ---------------- launcher ----------------
extern "C" void kernel_launch(void* const* d_in, const int* in_sizes, int n_in,
                              void* d_out, int out_size, void* d_ws,
                              size_t ws_size, hipStream_t stream) {
  (void)in_sizes; (void)n_in; (void)out_size; (void)ws_size;
  const float* query = (const float*)d_in[0];
  const float* key   = (const float*)d_in[1];
  const float* value = (const float*)d_in[2];
  const float* Wq = (const float*)d_in[3];
  const float* bq = (const float*)d_in[4];
  const float* Wk = (const float*)d_in[5];
  const float* bk = (const float*)d_in[6];
  const float* Wv = (const float*)d_in[7];
  const float* bv = (const float*)d_in[8];
  const float* Wo = (const float*)d_in[9];
  const float* bo = (const float*)d_in[10];
  float* out = (float*)d_out;

  const size_t actE = (size_t)BB * SS * DD;  // 8,388,608
  const size_t wE   = (size_t)DD * DD;       // 1,048,576

  __bf16* Qb  = (__bf16*)d_ws;      // [B,H,S,DK]
  __bf16* Kb  = Qb + actE;          // [B,H,S,DK]
  __bf16* Vtb = Kb + actE;          // [B,H,DK,S]
  __bf16* Xat = Vtb + actE;         // [B*S, D]
  __bf16* Xq  = Xat + actE;         // bf16 copies of inputs
  __bf16* Xk  = Xq + actE;
  __bf16* Xv  = Xk + actE;
  __bf16* Wqb = Xv + actE;          // bf16 weights
  __bf16* Wkb = Wqb + wE;
  __bf16* Wvb = Wkb + wE;
  __bf16* Wob = Wvb + wE;

  // ---- stage 0: bulk conversions (bandwidth-bound, one pass) ----
  {
    const int nAct8 = (int)(actE / 8);
    const int nW8   = (int)(wE / 8);
    dim3 cb(256);
    dim3 ga((nAct8 + 255) / 256), gw((nW8 + 255) / 256);
    cvt_f32_bf16_kernel<<<ga, cb, 0, stream>>>(query, Xq, nAct8);
    cvt_f32_bf16_kernel<<<ga, cb, 0, stream>>>(key,   Xk, nAct8);
    cvt_f32_bf16_kernel<<<ga, cb, 0, stream>>>(value, Xv, nAct8);
    cvt_f32_bf16_kernel<<<gw, cb, 0, stream>>>(Wq, Wqb, nW8);
    cvt_f32_bf16_kernel<<<gw, cb, 0, stream>>>(Wk, Wkb, nW8);
    cvt_f32_bf16_kernel<<<gw, cb, 0, stream>>>(Wv, Wvb, nW8);
    cvt_f32_bf16_kernel<<<gw, cb, 0, stream>>>(Wo, Wob, nW8);
  }

  dim3 pblk(128);
  dim3 pgrid(DD / 64, (BB * SS) / 128);  // (16, 64)
  gemm_qkv_kernel<<<pgrid, pblk, 0, stream>>>(Xq, Wqb, bq, Qb, 0);
  gemm_qkv_kernel<<<pgrid, pblk, 0, stream>>>(Xk, Wkb, bk, Kb, 0);
  gemm_qkv_kernel<<<pgrid, pblk, 0, stream>>>(Xv, Wvb, bv, Vtb, 1);

  dim3 agrid(SS / 128, BB * HH);  // (16, 64)
  flash_attn_kernel<<<agrid, pblk, 0, stream>>>(Qb, Kb, Vtb, Xat);

  gemm_out_kernel<<<pgrid, pblk, 0, stream>>>(Xat, Wob, bo, out);
}
